// mLSTMCell_2980707303872
// MI455X (gfx1250) — matrix operand order repaired and checked by reading, not compiled, not run on hardware
//
#include <hip/hip_runtime.h>

typedef __attribute__((ext_vector_type(16))) __bf16 v16bf;
typedef __attribute__((ext_vector_type(8)))  __bf16 v8bf;
typedef __attribute__((ext_vector_type(8)))  float  v8f;

#define BB   2
#define SS   2048
#define HH   1024
#define NHH  8
#define DHH  128
#define QK_SCALE 0.08838834764831845f   // 1/sqrt(128)
#define NEGBIG  -1e30f

// ---------------------------------------------------------------------------
// Kernel 1: convert q,k -> bf16 [B,NH,S,DH]; v -> bf16 transposed [B,NH,DH,S]
// ---------------------------------------------------------------------------
__global__ __launch_bounds__(256) void prep_qkv_kernel(
    const float* __restrict__ q, const float* __restrict__ k,
    const float* __restrict__ v,
    __bf16* __restrict__ Qb, __bf16* __restrict__ Kb, __bf16* __restrict__ Vt)
{
    size_t idx = (size_t)blockIdx.x * 256 + threadIdx.x;   // B*NH*S*DH
    int d = idx & 127;
    int s = (idx >> 7) & 2047;
    int h = (idx >> 18) & 7;
    int b = (int)(idx >> 21);
    size_t src = ((size_t)(b * SS + s)) * HH + h * DHH + d;
    Qb[idx] = (__bf16)q[src];
    Kb[idx] = (__bf16)k[src];
    Vt[(((size_t)(b * NHH + h) * DHH + d) * SS) + s] = (__bf16)v[src];
}

// ---------------------------------------------------------------------------
// Kernel 2: gates. One block per (b,s). 16 groups x 16 lanes reduce the
// 3072-wide dot products. IG[b,h,s] = ig ; LFG[b,h,s] = log_sigmoid(fg).
// ---------------------------------------------------------------------------
__global__ __launch_bounds__(256) void gates_kernel(
    const float* __restrict__ q, const float* __restrict__ k,
    const float* __restrict__ v,
    const float* __restrict__ w_i, const float* __restrict__ b_i,
    const float* __restrict__ w_f, const float* __restrict__ b_f,
    float* __restrict__ IG, float* __restrict__ LFG)
{
    int bs  = blockIdx.x;          // 0 .. B*S-1
    int b   = bs >> 11;
    int s   = bs & 2047;
    int tid = threadIdx.x;
    int grp = tid >> 4;            // 0..15
    int l16 = tid & 15;
    int g   = grp & 7;
    bool isF = grp >= 8;
    const float* w = isF ? w_f : w_i;
    const float* qrow = q + (size_t)bs * HH;
    const float* krow = k + (size_t)bs * HH;
    const float* vrow = v + (size_t)bs * HH;
    float sum = 0.f;
    for (int i = l16; i < HH; i += 16) {
        sum += qrow[i] * w[(size_t)i * NHH + g];
        sum += krow[i] * w[(size_t)(HH + i) * NHH + g];
        sum += vrow[i] * w[(size_t)(2 * HH + i) * NHH + g];
    }
    sum += __shfl_xor(sum, 8, 16);
    sum += __shfl_xor(sum, 4, 16);
    sum += __shfl_xor(sum, 2, 16);
    sum += __shfl_xor(sum, 1, 16);
    if (l16 == 0) {
        size_t o = ((size_t)(b * NHH + g)) * SS + s;
        if (!isF) {
            IG[o] = sum + b_i[g];
        } else {
            float z  = sum + b_f[g];
            float ls = fminf(z, 0.f) - log1pf(__expf(-fabsf(z)));
            LFG[o] = ls;
        }
    }
}

// ---------------------------------------------------------------------------
// Kernel 3: per (b,h), one wave32:
//   Fc[s]  = inclusive cumsum of log_sigmoid(fg)
//   CV[s]  = ig[s] - Fc[s]                  (column value of log_D)
//   PM[s]  = inclusive prefix max of CV     (row-max for all cols <= s)
// ---------------------------------------------------------------------------
__global__ __launch_bounds__(32) void scan_kernel(
    const float* __restrict__ LFG, const float* __restrict__ IG,
    float* __restrict__ Fc, float* __restrict__ CV, float* __restrict__ PM)
{
    int bh   = blockIdx.x;
    int lane = threadIdx.x;
    const float* src = LFG + (size_t)bh * SS;
    const float* gi  = IG + (size_t)bh * SS;
    float* fdst = Fc + (size_t)bh * SS;
    float* cdst = CV + (size_t)bh * SS;
    float* pdst = PM + (size_t)bh * SS;
    float carry = 0.f;
    float pcarry = NEGBIG;
    for (int c = 0; c < SS / 32; ++c) {
        int i = c * 32 + lane;
        float x = src[i];
        #pragma unroll
        for (int off = 1; off < 32; off <<= 1) {
            float y = __shfl_up(x, off);
            if (lane >= off) x += y;
        }
        x += carry;
        fdst[i] = x;
        float cv = gi[i] - x;
        cdst[i] = cv;
        float pm = cv;
        #pragma unroll
        for (int off = 1; off < 32; off <<= 1) {
            float y = __shfl_up(pm, off);
            if (lane >= off) pm = fmaxf(pm, y);
        }
        pm = fmaxf(pm, pcarry);
        pdst[i] = pm;
        carry  = __shfl(x, 31);
        pcarry = __shfl(pm, 31);
    }
}

// ---------------------------------------------------------------------------
// Kernel 4: flash mLSTM, fixed-max, 32-row M tile (two 16-row subtiles per
// wave). Each K/V B-fragment load feeds two WMMAs -> 34 WMMAs per 16 loads.
// ---------------------------------------------------------------------------
__global__ __launch_bounds__(32) void flash_mlstm_kernel(
    const __bf16* __restrict__ Qb, const __bf16* __restrict__ Kb,
    const __bf16* __restrict__ Vt, const float* __restrict__ Fc,
    const float* __restrict__ CV,  const float* __restrict__ PM,
    const float* __restrict__ rms_scale, float* __restrict__ out)
{
    __shared__ __align__(16) __bf16 Pbuf[2][16 * 32];  // two 16x32 bf16 tiles

    const int bid  = blockIdx.x;
    const int bh   = bid >> 6;          // S/32 = 64 row tiles per (b,h)
    const int rt   = bid & 63;
    const int r0   = rt * 32;           // 32-aligned -> single diagonal chunk
    const int b    = bh >> 3;
    const int h    = bh & 7;
    const int lane = threadIdx.x;
    const int hf   = lane >> 4;         // wave half: 0 or 1
    const int l16  = lane & 15;

    const __bf16* Qh = Qb + (size_t)bh * SS * DHH;
    const __bf16* Kh = Kb + (size_t)bh * SS * DHH;
    const __bf16* Vh = Vt + (size_t)bh * DHH * SS;
    const float*  Fh = Fc + (size_t)bh * SS;
    const float*  Ch = CV + (size_t)bh * SS;

    // --- Q A-fragments for both 16-row subtiles
    v16bf aq[2][4];
    #pragma unroll
    for (int mt = 0; mt < 2; ++mt) {
        const __bf16* qrow = Qh + (size_t)(r0 + mt * 16 + l16) * DHH + hf * 16;
        #pragma unroll
        for (int kc = 0; kc < 4; ++kc) {
            v8bf lo = *(const v8bf*)(qrow + kc * 32);
            v8bf hi = *(const v8bf*)(qrow + kc * 32 + 8);
            aq[mt][kc] = __builtin_shufflevector(lo, hi,
                         0, 1, 2, 3, 4, 5, 6, 7, 8, 9, 10, 11, 12, 13, 14, 15);
        }
    }

    // all-ones B fragment: folds the signed row-sum of P into extra WMMAs
    v16bf ones;
    #pragma unroll
    for (int i = 0; i < 16; ++i) ones[i] = (__bf16)1.0f;

    // per-lane row metadata
    float F_row[2][8];
    #pragma unroll
    for (int mt = 0; mt < 2; ++mt)
        #pragma unroll
        for (int j = 0; j < 8; ++j)
            F_row[mt][j] = Fh[r0 + mt * 16 + j + hf * 8];

    const int nFull = r0 >> 5;          // chunks with all 32 cols < r0
    const int tc0   = r0;               // diagonal tail chunk base

    // prefix max over all columns t < r0 (row-independent part of row max)
    const float Cpre = (r0 > 0) ? PM[(size_t)bh * SS + r0 - 1] : NEGBIG;

    // tail chunk column values (reused later)
    const float cvt0 = Ch[tc0 + l16];
    const float cvt1 = Ch[tc0 + 16 + l16];

    // fixed row max m = max(F_row + Cpre, masked max over tail chunk)
    float mrow[2][8], rowfacC[2][8];
    #pragma unroll
    for (int mt = 0; mt < 2; ++mt) {
        #pragma unroll
        for (int j = 0; j < 8; ++j) {
            const int row = r0 + mt * 16 + j + hf * 8;
            float ld0 = (tc0 + l16      <= row) ? (F_row[mt][j] + cvt0) : NEGBIG;
            float ld1 = (tc0 + 16 + l16 <= row) ? (F_row[mt][j] + cvt1) : NEGBIG;
            float rm = fmaxf(ld0, ld1);
            rm = fmaxf(rm, __shfl_xor(rm, 1));
            rm = fmaxf(rm, __shfl_xor(rm, 2));
            rm = fmaxf(rm, __shfl_xor(rm, 4));
            rm = fmaxf(rm, __shfl_xor(rm, 8));
            float mn = fmaxf(F_row[mt][j] + Cpre, rm);
            mrow[mt][j]    = mn;
            rowfacC[mt][j] = QK_SCALE * __expf(F_row[mt][j] + Cpre - mn);
        }
    }

    v8f acc[2][9];                      // per subtile: 8 DH subtiles + rowsum
    #pragma unroll
    for (int mt = 0; mt < 2; ++mt)
        #pragma unroll
        for (int d = 0; d < 9; ++d) acc[mt][d] = (v8f){};

    // ---------------- main loop over full (unmasked) chunks ----------------
    for (int ch = 0; ch < nFull; ++ch) {
        const int cc0 = ch * 32;
        __builtin_prefetch(Kh + (size_t)(cc0 + 32 + l16) * DHH, 0, 3);
        __builtin_prefetch(Vh + (size_t)l16 * SS + cc0 + 32, 0, 3);

        // S = Q * K^T: each bk fragment feeds both M subtiles (16 WMMAs)
        v8f sa[2][2] = {{(v8f){}, (v8f){}}, {(v8f){}, (v8f){}}};
        {
            const __bf16* kb0 = Kh + (size_t)(cc0 + l16) * DHH + hf * 16;
            const __bf16* kb1 = Kh + (size_t)(cc0 + 16 + l16) * DHH + hf * 16;
            #pragma unroll
            for (int kc = 0; kc < 4; ++kc) {
                v16bf bk0 = *(const v16bf*)(kb0 + kc * 32);
                v16bf bk1 = *(const v16bf*)(kb1 + kc * 32);
                sa[0][0] = __builtin_amdgcn_wmma_f32_16x16x32_bf16(
                        false, aq[0][kc], false, bk0, (short)0, sa[0][0], false, false);
                sa[0][1] = __builtin_amdgcn_wmma_f32_16x16x32_bf16(
                        false, aq[0][kc], false, bk1, (short)0, sa[0][1], false, false);
                sa[1][0] = __builtin_amdgcn_wmma_f32_16x16x32_bf16(
                        false, aq[1][kc], false, bk0, (short)0, sa[1][0], false, false);
                sa[1][1] = __builtin_amdgcn_wmma_f32_16x16x32_bf16(
                        false, aq[1][kc], false, bk1, (short)0, sa[1][1], false, false);
            }
        }

        // P = qk * rowfacC * exp(cv - Cpre); both factors <= 1, no masking
        const float cf0 = __expf(Ch[cc0 + l16]      - Cpre);
        const float cf1 = __expf(Ch[cc0 + 16 + l16] - Cpre);
        #pragma unroll
        for (int mt = 0; mt < 2; ++mt) {
            #pragma unroll
            for (int j = 0; j < 8; ++j) {
                float p0 = sa[mt][0][j] * rowfacC[mt][j] * cf0;
                float p1 = sa[mt][1][j] * rowfacC[mt][j] * cf1;
                Pbuf[mt][(j + hf * 8) * 32 + l16]      = (__bf16)p0;
                Pbuf[mt][(j + hf * 8) * 32 + 16 + l16] = (__bf16)p1;
            }
        }
        __syncthreads();

        // reload both P tiles as A-fragments via LDS
        v16bf pa[2];
        #pragma unroll
        for (int mt = 0; mt < 2; ++mt) {
            v8bf plo = *(const v8bf*)(&Pbuf[mt][l16 * 32 + hf * 16]);
            v8bf phi = *(const v8bf*)(&Pbuf[mt][l16 * 32 + hf * 16 + 8]);
            pa[mt] = __builtin_shufflevector(plo, phi,
                     0, 1, 2, 3, 4, 5, 6, 7, 8, 9, 10, 11, 12, 13, 14, 15);
        }

        // acc += P * [V | 1]: each bv fragment feeds both M subtiles (18 WMMAs)
        #pragma unroll
        for (int d = 0; d < 8; ++d) {
            v16bf bv = *(const v16bf*)(Vh + (size_t)(d * 16 + l16) * SS
                                          + cc0 + hf * 16);
            acc[0][d] = __builtin_amdgcn_wmma_f32_16x16x32_bf16(
                        false, pa[0], false, bv, (short)0, acc[0][d], false, false);
            acc[1][d] = __builtin_amdgcn_wmma_f32_16x16x32_bf16(
                        false, pa[1], false, bv, (short)0, acc[1][d], false, false);
        }
        acc[0][8] = __builtin_amdgcn_wmma_f32_16x16x32_bf16(
                    false, pa[0], false, ones, (short)0, acc[0][8], false, false);
        acc[1][8] = __builtin_amdgcn_wmma_f32_16x16x32_bf16(
                    false, pa[1], false, ones, (short)0, acc[1][8], false, false);
        __syncthreads();
    }

    // ---------------- diagonal tail chunk (masked, runs once) --------------
    {
        const int cc0 = tc0;
        v8f sa[2][2] = {{(v8f){}, (v8f){}}, {(v8f){}, (v8f){}}};
        {
            const __bf16* kb0 = Kh + (size_t)(cc0 + l16) * DHH + hf * 16;
            const __bf16* kb1 = Kh + (size_t)(cc0 + 16 + l16) * DHH + hf * 16;
            #pragma unroll
            for (int kc = 0; kc < 4; ++kc) {
                v16bf bk0 = *(const v16bf*)(kb0 + kc * 32);
                v16bf bk1 = *(const v16bf*)(kb1 + kc * 32);
                sa[0][0] = __builtin_amdgcn_wmma_f32_16x16x32_bf16(
                        false, aq[0][kc], false, bk0, (short)0, sa[0][0], false, false);
                sa[0][1] = __builtin_amdgcn_wmma_f32_16x16x32_bf16(
                        false, aq[0][kc], false, bk1, (short)0, sa[0][1], false, false);
                sa[1][0] = __builtin_amdgcn_wmma_f32_16x16x32_bf16(
                        false, aq[1][kc], false, bk0, (short)0, sa[1][0], false, false);
                sa[1][1] = __builtin_amdgcn_wmma_f32_16x16x32_bf16(
                        false, aq[1][kc], false, bk1, (short)0, sa[1][1], false, false);
            }
        }
        #pragma unroll
        for (int mt = 0; mt < 2; ++mt) {
            #pragma unroll
            for (int j = 0; j < 8; ++j) {
                const int row = r0 + mt * 16 + j + hf * 8;
                float er = F_row[mt][j] - mrow[mt][j];
                float e0 = (cc0 + l16      <= row) ? __expf(er + cvt0) : 0.f;
                float e1 = (cc0 + 16 + l16 <= row) ? __expf(er + cvt1) : 0.f;
                float p0 = sa[mt][0][j] * QK_SCALE * e0;
                float p1 = sa[mt][1][j] * QK_SCALE * e1;
                Pbuf[mt][(j + hf * 8) * 32 + l16]      = (__bf16)p0;
                Pbuf[mt][(j + hf * 8) * 32 + 16 + l16] = (__bf16)p1;
            }
        }
        __syncthreads();
        v16bf pa[2];
        #pragma unroll
        for (int mt = 0; mt < 2; ++mt) {
            v8bf plo = *(const v8bf*)(&Pbuf[mt][l16 * 32 + hf * 16]);
            v8bf phi = *(const v8bf*)(&Pbuf[mt][l16 * 32 + hf * 16 + 8]);
            pa[mt] = __builtin_shufflevector(plo, phi,
                     0, 1, 2, 3, 4, 5, 6, 7, 8, 9, 10, 11, 12, 13, 14, 15);
        }
        #pragma unroll
        for (int d = 0; d < 8; ++d) {
            v16bf bv = *(const v16bf*)(Vh + (size_t)(d * 16 + l16) * SS
                                          + cc0 + hf * 16);
            acc[0][d] = __builtin_amdgcn_wmma_f32_16x16x32_bf16(
                        false, pa[0], false, bv, (short)0, acc[0][d], false, false);
            acc[1][d] = __builtin_amdgcn_wmma_f32_16x16x32_bf16(
                        false, pa[1], false, bv, (short)0, acc[1][d], false, false);
        }
        acc[0][8] = __builtin_amdgcn_wmma_f32_16x16x32_bf16(
                    false, pa[0], false, ones, (short)0, acc[0][8], false, false);
        acc[1][8] = __builtin_amdgcn_wmma_f32_16x16x32_bf16(
                    false, pa[1], false, ones, (short)0, acc[1][8], false, false);
    }

    // ---------------- epilogue: normalizer, RMS norm, store ----------------
    #pragma unroll
    for (int mt = 0; mt < 2; ++mt) {
        float inv[8];
        #pragma unroll
        for (int j = 0; j < 8; ++j) {
            float bsum = acc[mt][8][j];      // uniform across lanes (ones B)
            float nrm = fmaxf(fabsf(bsum), __expf(-mrow[mt][j])) + 1e-6f;
            inv[j] = 1.f / nrm;
        }
        float ssq[8];
        #pragma unroll
        for (int j = 0; j < 8; ++j) ssq[j] = 0.f;
        #pragma unroll
        for (int d = 0; d < 8; ++d) {
            #pragma unroll
            for (int j = 0; j < 8; ++j) {
                float hv = acc[mt][d][j] * inv[j];
                acc[mt][d][j] = hv;
                ssq[j] += hv * hv;
            }
        }
        float rs[8];
        #pragma unroll
        for (int j = 0; j < 8; ++j) {
            float s2 = ssq[j];
            s2 += __shfl_xor(s2, 1);
            s2 += __shfl_xor(s2, 2);
            s2 += __shfl_xor(s2, 4);
            s2 += __shfl_xor(s2, 8);
            rs[j] = rsqrtf(s2 * (1.f / DHH) + 1e-6f);
        }
        #pragma unroll
        for (int d = 0; d < 8; ++d) {
            float scl = 1.f + rms_scale[d * 16 + l16];
            #pragma unroll
            for (int j = 0; j < 8; ++j) {
                size_t o = ((size_t)(b * SS + r0 + mt * 16 + j + hf * 8)) * HH
                         + h * DHH + d * 16 + l16;
                out[o] = acc[mt][d][j] * rs[j] * scl;
            }
        }
    }
}

// ---------------------------------------------------------------------------
// Launcher
// ---------------------------------------------------------------------------
extern "C" void kernel_launch(void* const* d_in, const int* in_sizes, int n_in,
                              void* d_out, int out_size, void* d_ws, size_t ws_size,
                              hipStream_t stream)
{
    const float* q   = (const float*)d_in[0];
    const float* k   = (const float*)d_in[1];
    const float* v   = (const float*)d_in[2];
    const float* w_i = (const float*)d_in[3];
    const float* b_i = (const float*)d_in[4];
    const float* w_f = (const float*)d_in[5];
    const float* b_f = (const float*)d_in[6];
    const float* rms = (const float*)d_in[7];
    float* out = (float*)d_out;

    const size_t QB = (size_t)BB * NHH * SS * DHH * sizeof(__bf16); // 8 MiB
    const size_t FB = (size_t)BB * NHH * SS * sizeof(float);        // 128 KiB
    char* ws = (char*)d_ws;
    __bf16* Qb  = (__bf16*)(ws);
    __bf16* Kb  = (__bf16*)(ws + QB);
    __bf16* Vt  = (__bf16*)(ws + 2 * QB);
    float*  Fc  = (float*)(ws + 3 * QB);
    float*  IG  = (float*)(ws + 3 * QB + FB);
    float*  LFG = (float*)(ws + 3 * QB + 2 * FB);
    float*  CVa = (float*)(ws + 3 * QB + 3 * FB);
    float*  PMa = (float*)(ws + 3 * QB + 4 * FB);

    const int totalQ = BB * NHH * SS * DHH;
    prep_qkv_kernel<<<totalQ / 256, 256, 0, stream>>>(q, k, v, Qb, Kb, Vt);
    gates_kernel<<<BB * SS, 256, 0, stream>>>(q, k, v, w_i, b_i, w_f, b_f, IG, LFG);
    scan_kernel<<<BB * NHH, 32, 0, stream>>>(LFG, IG, Fc, CVa, PMa);
    flash_mlstm_kernel<<<BB * NHH * (SS / 32), 32, 0, stream>>>(
        Qb, Kb, Vt, Fc, CVa, PMa, rms, out);
}
